// MultiheadSelfAttention_55070070669766
// MI455X (gfx1250) — matrix operand order repaired and checked by reading, compiled, not run on hardware
//
#include <hip/hip_runtime.h>

// ---------------------------------------------------------------------------
// MI455X / gfx1250 multi-head self-attention with RoPE, bf16 WMMA everywhere.
//   B=2, S=2048, D_MODEL=2048, H=16, D_HEAD=128
// Pipeline:
//   1) cast x, Wq, Wk, Wv, Wo  fp32 -> bf16
//   2) Q/K/V = x @ W^T + b        (WMMA GEMM, double-buffered TDM staging)
//   3) RoPE + head split (Q,K -> [b][h][s][d]); V -> transposed [b][h][d][s]
//   4) flash-attention per (b, h, q-tile): double-buffered TDM K/V tiles,
//      WMMA QK^T + PV, online softmax
//   5) out = attn @ Wo^T + bo     (WMMA GEMM, fp32 out)
// ---------------------------------------------------------------------------

typedef __bf16 bf16;
typedef __attribute__((ext_vector_type(16))) __bf16 v16bf;
typedef __attribute__((ext_vector_type(8)))  float  v8f;
typedef __attribute__((ext_vector_type(4)))  unsigned tdm_u32x4;
typedef __attribute__((ext_vector_type(8)))  int      tdm_i32x8;
typedef __attribute__((ext_vector_type(4)))  int      tdm_i32x4;

#define BATCH 2
#define SEQ   2048
#define DMODEL 2048
#define NHEAD 16
#define DHEAD 128
#define MROWS (BATCH * SEQ)        // 4096
#define ATT_SCALE 0.08838834764831845f   // 1/sqrt(128)
#define NEG_INF_F (-1.0e30f)

union FragU {
  uint4 u4[2];
  v16bf v;
};

__device__ __forceinline__ v8f v8f_zero() {
  v8f z = {0.f, 0.f, 0.f, 0.f, 0.f, 0.f, 0.f, 0.f};
  return z;
}

__device__ __forceinline__ unsigned lds_addr32(const void* p) {
  return (unsigned)(unsigned long long)(uintptr_t)p;  // LDS offset = flat[31:0]
}

// ------------------------- Tensor Data Mover (TDM) -------------------------
// 2-D tile DMA: global -> LDS, bf16 elements (data_size = 2B).
// D# layout per CDNA5 ISA ch.8 (group0 128b, group1 256b, groups 2/3 zero).
union TdmG0 { unsigned u[4];            tdm_u32x4 v; };
union TdmG1 { unsigned long long q[4];  tdm_i32x8 v; };

__device__ __forceinline__ void tdm_load_2d_bf16(
    unsigned lds_byte_addr, const void* gaddr,
    unsigned tensor_d0, unsigned tensor_d1, unsigned long long stride0_elts,
    unsigned tile_d0, unsigned tile_d1) {
  const unsigned long long ga = (unsigned long long)gaddr;
  TdmG0 g0;
  g0.u[0] = 1u;                                      // count=1, user descriptor
  g0.u[1] = lds_byte_addr;                           // bits 63:32  lds_addr
  g0.u[2] = (unsigned)(ga & 0xffffffffull);          // global_addr[31:0]
  g0.u[3] = (unsigned)((ga >> 32) & 0x1ffffffull)    // global_addr[56:32]
          | (2u << 30);                              // type = 2 ("image")
  TdmG1 g1;
  g1.q[0] = (1ull << 16)                                             // data_size=2B
          | ((unsigned long long)(tensor_d0 & 0xffffu) << 48);       // tensor_dim0[15:0]
  g1.q[1] = ((unsigned long long)(tensor_d0 >> 16) & 0xffffull)      // tensor_dim0[31:16]
          | ((unsigned long long)tensor_d1 << 16)                    // tensor_dim1
          | ((unsigned long long)(tile_d0 & 0xffffu) << 48);         // tile_dim0
  g1.q[2] = (unsigned long long)(tile_d1 & 0xffffu)                  // tile_dim1
          | ((stride0_elts & 0xffffffffull) << 32);                  // dim0_stride[31:0]
  g1.q[3] = (stride0_elts >> 32) & 0xffffull;                        // dim0_stride[47:32]
  tdm_i32x4 z4 = {0, 0, 0, 0};
#if __clang_major__ >= 23
  tdm_i32x8 z8 = {0, 0, 0, 0, 0, 0, 0, 0};
  __builtin_amdgcn_tensor_load_to_lds(g0.v, g1.v, z4, z4, z8, 0);
#else
  __builtin_amdgcn_tensor_load_to_lds(g0.v, g1.v, z4, z4, 0);
#endif
}

// A-fragment (16x32 bf16, M x K). ISA layout:
//   lanes 0-15 : M=lane,    K = {0..7, 16..23}
//   lanes 16-31: M=lane-16, K = {8..15, 24..31}
__device__ __forceinline__ v16bf frag_a_lds(const bf16* t, int stride, int row0, int lane) {
  int lm = lane & 15;
  int kb = (lane & 16) ? 8 : 0;
  const bf16* p = t + (size_t)(row0 + lm) * stride;
  FragU f;
  f.u4[0] = *reinterpret_cast<const uint4*>(p + kb);
  f.u4[1] = *reinterpret_cast<const uint4*>(p + kb + 16);
  return f.v;
}

// B-fragment (32x16 bf16, K x N), stored N-major in memory (row = N, 32 K's).
//   lanes 0-15 : N=lane,    K = {0..7, 16..23}
//   lanes 16-31: N=lane-16, K = {8..15, 24..31}
__device__ __forceinline__ v16bf frag_b_lds(const bf16* t, int stride, int row0, int lane) {
  int lm = lane & 15;
  int kb = (lane & 16) ? 8 : 0;
  const bf16* p = t + (size_t)(row0 + lm) * stride;
  FragU f;
  f.u4[0] = *reinterpret_cast<const uint4*>(p + kb);
  f.u4[1] = *reinterpret_cast<const uint4*>(p + kb + 16);
  return f.v;
}

// --------------------------- cast fp32 -> bf16 -----------------------------
__global__ void cast_f32_to_bf16(const float* __restrict__ src,
                                 bf16* __restrict__ dst, long n) {
  long i = (long)blockIdx.x * blockDim.x + threadIdx.x;
  if (i < n) dst[i] = (bf16)src[i];
}

// --------------------------- generic WMMA GEMM -----------------------------
// C[M x N] = A[M x K] * Bm[N x K]^T + bias[N]
// Block: 256 threads = 8 waves; block tile 128(M) x 128(N); K-step 32.
// Tiles staged by the Tensor Data Mover, double-buffered so the DMA of
// K-slab i+1 overlaps the WMMAs on K-slab i.
template <bool STORE_F32>
__global__ __launch_bounds__(256) void gemm_wmma_bf16(
    const bf16* __restrict__ A, const bf16* __restrict__ Bm,
    const float* __restrict__ bias, void* __restrict__ Cout,
    int Mdim, int Ndim, int Kdim) {
  __shared__ bf16 As[2][128 * 32];
  __shared__ bf16 Bs[2][128 * 32];

  const int tid = threadIdx.x;
  const int wave = tid >> 5;
  const int lane = tid & 31;
  const int wm = wave >> 1;   // 0..3 -> 32-row group
  const int wn = wave & 1;    // 0..1 -> 64-col group
  const int rowA0 = blockIdx.x * 128;
  const int colB0 = blockIdx.y * 128;

  v8f acc[2][4];
#pragma unroll
  for (int i = 0; i < 2; ++i)
#pragma unroll
    for (int j = 0; j < 4; ++j) acc[i][j] = v8f_zero();

  const unsigned asAddr[2] = {lds_addr32(&As[0][0]), lds_addr32(&As[1][0])};
  const unsigned bsAddr[2] = {lds_addr32(&Bs[0][0]), lds_addr32(&Bs[1][0])};
  const int ldr = tid >> 1;          // used only for prefetch addressing
  const int ldc = (tid & 1) * 16;

  const int nk = Kdim >> 5;
  if (wave == 0) {  // prologue: DMA K-slab 0 into buffer 0
    tdm_load_2d_bf16(asAddr[0], &A[(size_t)rowA0 * Kdim], Kdim, Mdim, Kdim, 32, 128);
    tdm_load_2d_bf16(bsAddr[0], &Bm[(size_t)colB0 * Kdim], Kdim, Ndim, Kdim, 32, 128);
  }

  for (int i = 0; i < nk; ++i) {
    const int buf = i & 1;
    __syncthreads();  // everyone done reading buf^1 before it is overwritten
    if (wave == 0) {
      if (i + 1 < nk) {
        const int kn = (i + 1) << 5;
        tdm_load_2d_bf16(asAddr[buf ^ 1], &A[(size_t)rowA0 * Kdim + kn],
                         Kdim, Mdim, Kdim, 32, 128);
        tdm_load_2d_bf16(bsAddr[buf ^ 1], &Bm[(size_t)colB0 * Kdim + kn],
                         Kdim, Ndim, Kdim, 32, 128);
        __builtin_amdgcn_s_wait_tensorcnt(2);  // slab i landed; i+1 in flight
      } else {
        __builtin_amdgcn_s_wait_tensorcnt(0);
      }
    }
    if (i + 2 < nk) {  // nudge slab i+2 toward L2 (global_prefetch_b8)
      const int kp = (i + 2) << 5;
      __builtin_prefetch(&A[(size_t)(rowA0 + ldr) * Kdim + kp + ldc], 0, 1);
      __builtin_prefetch(&Bm[(size_t)(colB0 + ldr) * Kdim + kp + ldc], 0, 1);
    }
    __syncthreads();  // slab i visible to all 8 waves

    v16bf af[2], bfg[4];
#pragma unroll
    for (int ii = 0; ii < 2; ++ii) af[ii] = frag_a_lds(As[buf], 32, wm * 32 + ii * 16, lane);
#pragma unroll
    for (int j = 0; j < 4; ++j) bfg[j] = frag_b_lds(Bs[buf], 32, wn * 64 + j * 16, lane);
#pragma unroll
    for (int ii = 0; ii < 2; ++ii)
#pragma unroll
      for (int j = 0; j < 4; ++j)
        acc[ii][j] = __builtin_amdgcn_wmma_f32_16x16x32_bf16(
            false, af[ii], false, bfg[j], (short)0, acc[ii][j], false, false);
  }

  // C fragment layout: lane l, vgpr r -> M = r + (l>=16 ? 8 : 0), N = l&15
  const int hs = lane >> 4;
  const int lm = lane & 15;
#pragma unroll
  for (int i = 0; i < 2; ++i)
#pragma unroll
    for (int j = 0; j < 4; ++j) {
      const int col = colB0 + wn * 64 + j * 16 + lm;
      const float bv = bias[col];
#pragma unroll
      for (int r = 0; r < 8; ++r) {
        const int row = rowA0 + wm * 32 + i * 16 + r + hs * 8;
        const float vv = acc[i][j][r] + bv;
        if (STORE_F32)
          reinterpret_cast<float*>(Cout)[(size_t)row * Ndim + col] = vv;
        else
          reinterpret_cast<bf16*>(Cout)[(size_t)row * Ndim + col] = (bf16)vv;
      }
    }
}

// ---------------- RoPE + split heads: [b*s][D] -> [b][h][s][dh] ------------
__global__ void rope_split_kernel(const bf16* __restrict__ src,
                                  bf16* __restrict__ dst) {
  const long idx = (long)blockIdx.x * blockDim.x + threadIdx.x;  // pair index
  const long npairs = (long)MROWS * (DMODEL / 2);
  if (idx >= npairs) return;
  const int f = (int)(idx & (DMODEL / 2 - 1)) * 2;  // even feature in d_model
  const long row = idx >> 10;                       // b*SEQ + s
  const int s = (int)(row & (SEQ - 1));
  const int b = (int)(row >> 11);
  const int h = f >> 7;            // head
  const int hl = f & (DHEAD - 1);  // 2*i within head (even)
  const float freq = __expf(-((float)hl / (float)DHEAD) * __logf(10000.0f));
  const float ang = (float)s * freq;
  const float c = __cosf(ang), sn = __sinf(ang);
  const float x1 = (float)src[row * DMODEL + f];
  const float x2 = (float)src[row * DMODEL + f + 1];
  const size_t base = (((size_t)(b * NHEAD + h) * SEQ + s) * DHEAD) + hl;
  dst[base] = (bf16)(x1 * c - x2 * sn);
  dst[base + 1] = (bf16)(x1 * sn + x2 * c);
}

// ------------- V split + transpose: [b*s][D] -> [b][h][dh][s] --------------
__global__ void split_v_kernel(const bf16* __restrict__ src,
                               bf16* __restrict__ dst) {
  const long idx = (long)blockIdx.x * blockDim.x + threadIdx.x;
  const long n = (long)MROWS * DMODEL;
  if (idx >= n) return;
  const int col = (int)(idx & (DMODEL - 1));
  const long row = idx >> 11;
  const int s = (int)(row & (SEQ - 1));
  const int b = (int)(row >> 11);
  const int h = col >> 7;
  const int d = col & (DHEAD - 1);
  dst[((size_t)(b * NHEAD + h) * DHEAD + d) * SEQ + s] = src[idx];
}

// ----------------------- flash attention (causal) --------------------------
// grid.x = q-tile (SEQ/128), grid.y = b*NHEAD + h; 256 threads = 8 waves.
// Double-buffered TDM stages key-tile i+1 while the 8 waves run WMMAs on
// key-tile i (wave 0 owns the descriptors + TENSORcnt).
__global__ __launch_bounds__(256) void attn_kernel(
    const bf16* __restrict__ Q,   // [b][h][s][dh]
    const bf16* __restrict__ K,   // [b][h][s][dh]
    const bf16* __restrict__ Vt,  // [b][h][dh][s]
    bf16* __restrict__ O) {       // [b][s][h*dh]
  __shared__ bf16 Ks[2][32 * 128];     // [key][d]
  __shared__ bf16 Vs[2][128 * 32];     // [d][key]
  __shared__ bf16 Ps[8][16 * 32];      // per-wave P scratch [row][key]

  const int bh = blockIdx.y;
  const int qt = blockIdx.x;
  const int tid = threadIdx.x;
  const int wave = tid >> 5;
  const int lane = tid & 31;
  const int lm = lane & 15;
  const int hs = lane >> 4;

  const bf16* Qbh = Q + (size_t)bh * SEQ * DHEAD;
  const bf16* Kbh = K + (size_t)bh * SEQ * DHEAD;
  const bf16* Vbh = Vt + (size_t)bh * DHEAD * SEQ;
  const int q0 = qt * 128 + wave * 16;  // wave's first q row

  const unsigned ksAddr[2] = {lds_addr32(&Ks[0][0]), lds_addr32(&Ks[1][0])};
  const unsigned vsAddr[2] = {lds_addr32(&Vs[0][0]), lds_addr32(&Vs[1][0])};

  // Q A-fragments: 4 K-steps over d (loaded once, reused for all key tiles)
  v16bf qf[4];
#pragma unroll
  for (int kk = 0; kk < 4; ++kk) {
    const bf16* p = Qbh + (size_t)(q0 + lm) * DHEAD + kk * 32 + (hs ? 8 : 0);
    FragU f;
    f.u4[0] = *reinterpret_cast<const uint4*>(p);
    f.u4[1] = *reinterpret_cast<const uint4*>(p + 16);
    qf[kk] = f.v;
  }

  v8f acc[8];
#pragma unroll
  for (int j = 0; j < 8; ++j) acc[j] = v8f_zero();
  float mrun[8], lrun[8];
#pragma unroll
  for (int r = 0; r < 8; ++r) { mrun[r] = NEG_INF_F; lrun[r] = 0.f; }

  const int nkt = qt * 4 + 4;  // key tiles (32 keys each), causal upper bound
  if (wave == 0) {  // prologue: DMA key-tile 0 into buffer 0
    tdm_load_2d_bf16(ksAddr[0], Kbh, DHEAD, SEQ, DHEAD, DHEAD, 32);
    tdm_load_2d_bf16(vsAddr[0], Vbh, SEQ, DHEAD, SEQ, 32, DHEAD);
  }

  for (int i = 0; i < nkt; ++i) {
    const int buf = i & 1;
    const int k0 = i << 5;
    __syncthreads();  // everyone done reading buf^1 before it is overwritten
    if (wave == 0) {
      if (i + 1 < nkt) {
        const int kn = (i + 1) << 5;
        tdm_load_2d_bf16(ksAddr[buf ^ 1], Kbh + (size_t)kn * DHEAD,
                         DHEAD, SEQ, DHEAD, DHEAD, 32);
        tdm_load_2d_bf16(vsAddr[buf ^ 1], Vbh + kn,
                         SEQ, DHEAD, SEQ, 32, DHEAD);
        __builtin_amdgcn_s_wait_tensorcnt(2);  // tile i landed; i+1 in flight
      } else {
        __builtin_amdgcn_s_wait_tensorcnt(0);
      }
    }
    __syncthreads();  // tile i visible to all 8 waves

    // S = Q K^T  (16 q x 32 keys): 2 N-tiles x 4 K-steps
    v8f sf[2];
    sf[0] = v8f_zero();
    sf[1] = v8f_zero();
#pragma unroll
    for (int nt = 0; nt < 2; ++nt)
#pragma unroll
      for (int kk = 0; kk < 4; ++kk) {
        v16bf bfg = frag_b_lds(Ks[buf] + kk * 32, 128, nt * 16, lane);
        sf[nt] = __builtin_amdgcn_wmma_f32_16x16x32_bf16(
            false, qf[kk], false, bfg, (short)0, sf[nt], false, false);
      }

    // scale + causal mask
#pragma unroll
    for (int nt = 0; nt < 2; ++nt)
#pragma unroll
      for (int r = 0; r < 8; ++r) {
        float s = sf[nt][r] * ATT_SCALE;
        const int kg = k0 + nt * 16 + lm;
        const int qg = q0 + r + hs * 8;
        sf[nt][r] = (kg > qg) ? NEG_INF_F : s;
      }

    // online softmax: row max / exp / row sum (reduce across 16-lane halves)
    float alpha[8];
#pragma unroll
    for (int r = 0; r < 8; ++r) {
      float v = fmaxf(sf[0][r], sf[1][r]);
#pragma unroll
      for (int off = 1; off < 16; off <<= 1) v = fmaxf(v, __shfl_xor(v, off));
      const float mnew = fmaxf(mrun[r], v);
      alpha[r] = __expf(mrun[r] - mnew);
      mrun[r] = mnew;
    }
#pragma unroll
    for (int nt = 0; nt < 2; ++nt)
#pragma unroll
      for (int r = 0; r < 8; ++r) sf[nt][r] = __expf(sf[nt][r] - mrun[r]);
#pragma unroll
    for (int r = 0; r < 8; ++r) {
      float v = sf[0][r] + sf[1][r];
#pragma unroll
      for (int off = 1; off < 16; off <<= 1) v += __shfl_xor(v, off);
      lrun[r] = lrun[r] * alpha[r] + v;
    }
#pragma unroll
    for (int j = 0; j < 8; ++j)
#pragma unroll
      for (int r = 0; r < 8; ++r) acc[j][r] *= alpha[r];

    // re-layout P (C-frag) -> A-frag via per-wave LDS bounce
    bf16* Pw = &Ps[wave][0];
#pragma unroll
    for (int nt = 0; nt < 2; ++nt)
#pragma unroll
      for (int r = 0; r < 8; ++r)
        Pw[(r + hs * 8) * 32 + nt * 16 + lm] = (bf16)sf[nt][r];
    asm volatile("s_wait_dscnt 0" ::: "memory");  // DS in-order per wave; belt & braces
    const v16bf pf = frag_a_lds(Pw, 32, 0, lane);

    // O += P * V  (A: 16x32 probs, B: 32 keys x 16 d per tile)
#pragma unroll
    for (int j = 0; j < 8; ++j) {
      v16bf bfg = frag_b_lds(Vs[buf], 32, j * 16, lane);
      acc[j] = __builtin_amdgcn_wmma_f32_16x16x32_bf16(
          false, pf, false, bfg, (short)0, acc[j], false, false);
    }
  }

  // normalize and store: O[b][q][h*128 + d]
  const int b = bh >> 4, h = bh & 15;
  float inv[8];
#pragma unroll
  for (int r = 0; r < 8; ++r) inv[r] = 1.0f / lrun[r];
#pragma unroll
  for (int j = 0; j < 8; ++j)
#pragma unroll
    for (int r = 0; r < 8; ++r) {
      const int q = q0 + r + hs * 8;
      const int d = j * 16 + lm;
      O[((size_t)b * SEQ + q) * DMODEL + h * DHEAD + d] = (bf16)(acc[j][r] * inv[r]);
    }
}

// ---------------------------------------------------------------------------
extern "C" void kernel_launch(void* const* d_in, const int* in_sizes, int n_in,
                              void* d_out, int out_size, void* d_ws, size_t ws_size,
                              hipStream_t stream) {
  const float* x  = (const float*)d_in[0];
  const float* Wq = (const float*)d_in[1];
  const float* bq = (const float*)d_in[2];
  const float* Wk = (const float*)d_in[3];
  const float* bk = (const float*)d_in[4];
  const float* Wv = (const float*)d_in[5];
  const float* bv = (const float*)d_in[6];
  const float* Wo = (const float*)d_in[7];
  const float* bo = (const float*)d_in[8];

  const size_t XE = (size_t)MROWS * DMODEL;   // 8M elements
  const size_t WE = (size_t)DMODEL * DMODEL;  // 4M elements

  char* w = (char*)d_ws;
  bf16* xb   = (bf16*)w;              w += XE * sizeof(bf16);
  bf16* wqb  = (bf16*)w;              w += WE * sizeof(bf16);
  bf16* wkb  = (bf16*)w;              w += WE * sizeof(bf16);
  bf16* wvb  = (bf16*)w;              w += WE * sizeof(bf16);
  bf16* wob  = (bf16*)w;              w += WE * sizeof(bf16);
  bf16* qraw = (bf16*)w;              w += XE * sizeof(bf16);
  bf16* kraw = (bf16*)w;              w += XE * sizeof(bf16);
  bf16* vraw = (bf16*)w;              w += XE * sizeof(bf16);
  bf16* qh   = (bf16*)w;              w += XE * sizeof(bf16);
  bf16* kh   = (bf16*)w;              w += XE * sizeof(bf16);
  bf16* vt   = xb;    // x dead after projections
  bf16* ao   = qraw;  // q-raw dead after rope

  const int thr = 256;
  // 1) casts
  cast_f32_to_bf16<<<(int)((XE + thr - 1) / thr), thr, 0, stream>>>(x, xb, (long)XE);
  cast_f32_to_bf16<<<(int)((WE + thr - 1) / thr), thr, 0, stream>>>(Wq, wqb, (long)WE);
  cast_f32_to_bf16<<<(int)((WE + thr - 1) / thr), thr, 0, stream>>>(Wk, wkb, (long)WE);
  cast_f32_to_bf16<<<(int)((WE + thr - 1) / thr), thr, 0, stream>>>(Wv, wvb, (long)WE);
  cast_f32_to_bf16<<<(int)((WE + thr - 1) / thr), thr, 0, stream>>>(Wo, wob, (long)WE);

  // 2) projections (bf16 out)
  dim3 ggrid(MROWS / 128, DMODEL / 128);
  gemm_wmma_bf16<false><<<ggrid, thr, 0, stream>>>(xb, wqb, bq, qraw, MROWS, DMODEL, DMODEL);
  gemm_wmma_bf16<false><<<ggrid, thr, 0, stream>>>(xb, wkb, bk, kraw, MROWS, DMODEL, DMODEL);
  gemm_wmma_bf16<false><<<ggrid, thr, 0, stream>>>(xb, wvb, bv, vraw, MROWS, DMODEL, DMODEL);

  // 3) RoPE + layout
  const long npairs = (long)MROWS * (DMODEL / 2);
  rope_split_kernel<<<(int)((npairs + thr - 1) / thr), thr, 0, stream>>>(qraw, qh);
  rope_split_kernel<<<(int)((npairs + thr - 1) / thr), thr, 0, stream>>>(kraw, kh);
  split_v_kernel<<<(int)((XE + thr - 1) / thr), thr, 0, stream>>>(vraw, vt);

  // 4) attention (double-buffered TDM K/V tiles)
  dim3 agrid(SEQ / 128, BATCH * NHEAD);
  attn_kernel<<<agrid, thr, 0, stream>>>(qh, kh, vt, ao);

  // 5) output projection (fp32 out)
  gemm_wmma_bf16<true><<<ggrid, thr, 0, stream>>>(ao, wob, bo, d_out, MROWS, DMODEL, DMODEL);

  (void)in_sizes; (void)n_in; (void)out_size; (void)ws_size;
}
